// CantorGlobalAttentionO_n_17970143167306
// MI455X (gfx1250) — matrix operand order repaired and checked by reading, not compiled, not run on hardware
//
#include <hip/hip_runtime.h>

typedef __attribute__((ext_vector_type(16))) __bf16 v16bf;
typedef __attribute__((ext_vector_type(8)))  float  v8f;

static constexpr int Bv  = 2;
static constexpr int Sv  = 2048;
static constexpr int Dv  = 1024;
static constexpr int Hv  = 16;
static constexpr int HDv = 64;
static constexpr int Kr  = 64;
static constexpr int TDQ = 3 * Dv;   // 3072

// ---------- bf16 helpers (bit-level, no arithmetic on __bf16) ----------
__device__ __forceinline__ unsigned short f2bf(float f) {
    unsigned int u = __float_as_uint(f);
    u += 0x7fffu + ((u >> 16) & 1u);          // round-to-nearest-even
    return (unsigned short)(u >> 16);
}
__device__ __forceinline__ float bf2f(unsigned short s) {
    return __uint_as_float(((unsigned int)s) << 16);
}

union FragU { uint4 q[2]; v16bf v; };
union U4S   { uint4 q; unsigned short s[8]; };

__device__ __forceinline__ v16bf ld_frag(const unsigned short* p0,
                                         const unsigned short* p1) {
    FragU f;
    f.q[0] = *(const uint4*)p0;
    f.q[1] = *(const uint4*)p1;
    return f.v;
}

// ---------- CDNA5 async global->LDS copy (ASYNCcnt-tracked) ----------
__device__ __forceinline__ void async_copy_b128(unsigned ldsOff,
                                                const unsigned short* gptr) {
    asm volatile("global_load_async_to_lds_b128 %0, %1, off"
                 :: "v"(ldsOff), "v"(gptr)
                 : "memory");
}
__device__ __forceinline__ void wait_async0() {
    asm volatile("s_wait_asynccnt 0x0" ::: "memory");
}

// ---------- fp32 -> bf16 conversion ----------
__global__ void cvt_f32_bf16(const float* __restrict__ src,
                             unsigned short* __restrict__ dst, int n) {
    int i = blockIdx.x * blockDim.x + threadIdx.x;
    if (i < n) dst[i] = f2bf(src[i]);
}

// ---------- bf16 WMMA GEMM:  C[M,N] = A[M,K] * B[K,N] + bias[N] ----------
// Block: 256 threads (8 waves), 128x128 macro tile. Wave = 64x32 (4x2 WMMA tiles).
// Double-buffered LDS; A tile via global_load_async_to_lds_b128, B tile via
// register pipeline (needs transpose into [n][k]).
template <bool BF16OUT>
__global__ __launch_bounds__(256) void gemm_bf16_wmma(
    const unsigned short* __restrict__ A,
    const unsigned short* __restrict__ Bm,
    const float* __restrict__ bias,
    void* __restrict__ outp,
    int N, int Kd) {
    __shared__ unsigned short As[2][128 * 40];   // [m][k], stride 40 halves
    __shared__ unsigned short Bs[2][128 * 40];   // [n][k], stride 40 halves

    const int tid  = threadIdx.x;
    const int m0   = blockIdx.y * 128;
    const int n0   = blockIdx.x * 128;
    const int lane = tid & 31;
    const int wave = tid >> 5;
    const int hlf  = lane >> 4;     // 0: lanes 0-15, 1: lanes 16-31
    const int rr   = lane & 15;
    const int wm   = wave >> 2;     // 0..1  -> 64 rows each
    const int wn   = wave & 3;      // 0..3  -> 32 cols each

    // staging assignments: 2 x 16B chunks per thread for each of A and B
    const int c0  = tid, c1 = tid + 256;
    const int am0 = c0 >> 2, ak0 = (c0 & 3) << 3;
    const int am1 = c1 >> 2, ak1 = (c1 & 3) << 3;
    const int bk0 = c0 >> 4, bn0 = (c0 & 15) << 3;
    const int bk1 = c1 >> 4, bn1 = (c1 & 15) << 3;

    const unsigned asLds    = (unsigned)(uintptr_t)(&As[0][0]);
    const unsigned asStride = 128u * 40u * 2u;   // bytes per A buffer

    v8f zero = {0.f, 0.f, 0.f, 0.f, 0.f, 0.f, 0.f, 0.f};
    v8f acc[4][2];
#pragma unroll
    for (int i = 0; i < 4; i++)
#pragma unroll
        for (int j = 0; j < 2; j++) acc[i][j] = zero;

    // ---- prologue: stage chunk 0 into buffer 0 ----
    async_copy_b128(asLds + (am0 * 40 + ak0) * 2, A + (size_t)(m0 + am0) * Kd + ak0);
    async_copy_b128(asLds + (am1 * 40 + ak1) * 2, A + (size_t)(m0 + am1) * Kd + ak1);
    {
        U4S u0, u1;
        u0.q = *(const uint4*)(Bm + (size_t)bk0 * N + n0 + bn0);
        u1.q = *(const uint4*)(Bm + (size_t)bk1 * N + n0 + bn1);
#pragma unroll
        for (int j = 0; j < 8; j++) {
            Bs[0][(bn0 + j) * 40 + bk0] = u0.s[j];
            Bs[0][(bn1 + j) * 40 + bk1] = u1.s[j];
        }
    }
    wait_async0();
    __syncthreads();

    int buf = 0;
    for (int k0 = 0; k0 < Kd; k0 += 32) {
        const bool notlast = (k0 + 32 < Kd);
        U4S u0, u1;
        if (notlast) {
            const int kn = k0 + 32;
            const unsigned dst = asLds + (unsigned)(buf ^ 1) * asStride;
            async_copy_b128(dst + (am0 * 40 + ak0) * 2,
                            A + (size_t)(m0 + am0) * Kd + kn + ak0);
            async_copy_b128(dst + (am1 * 40 + ak1) * 2,
                            A + (size_t)(m0 + am1) * Kd + kn + ak1);
            u0.q = *(const uint4*)(Bm + (size_t)(kn + bk0) * N + n0 + bn0);
            u1.q = *(const uint4*)(Bm + (size_t)(kn + bk1) * N + n0 + bn1);
        }

        // A fragment 16x32 bf16: lane<16 row=rr K{0..7,16..23}; lane>=16 K{8..15,24..31}
        v16bf af[4];
#pragma unroll
        for (int i = 0; i < 4; i++) {
            const unsigned short* base = &As[buf][(wm * 64 + i * 16 + rr) * 40];
            af[i] = ld_frag(base + hlf * 8, base + 16 + hlf * 8);
        }
        // B fragment 32x16 bf16: lane<16 col=rr K{0..7,16..23}; lane>=16 K{8..15,24..31}
        v16bf bfr[2];
#pragma unroll
        for (int j = 0; j < 2; j++) {
            const unsigned short* base = &Bs[buf][(wn * 32 + j * 16 + rr) * 40];
            bfr[j] = ld_frag(base + hlf * 8, base + 16 + hlf * 8);
        }
#pragma unroll
        for (int i = 0; i < 4; i++)
#pragma unroll
            for (int j = 0; j < 2; j++)
                acc[i][j] = __builtin_amdgcn_wmma_f32_16x16x32_bf16(
                    false, af[i], false, bfr[j], (short)0, acc[i][j], false, false);

        if (notlast) {
            // transpose-store next B tile behind the WMMA block
#pragma unroll
            for (int j = 0; j < 8; j++) {
                Bs[buf ^ 1][(bn0 + j) * 40 + bk0] = u0.s[j];
                Bs[buf ^ 1][(bn1 + j) * 40 + bk1] = u1.s[j];
            }
            wait_async0();
        }
        __syncthreads();
        buf ^= 1;
    }

    // epilogue: C/D layout: col = lane&15, row = e (+8 for lanes>=16)
#pragma unroll
    for (int i = 0; i < 4; i++) {
#pragma unroll
        for (int j = 0; j < 2; j++) {
            const int col  = n0 + wn * 32 + j * 16 + rr;
            const float bv = bias[col];
            const int rowb = m0 + wm * 64 + i * 16 + hlf * 8;
#pragma unroll
            for (int e = 0; e < 8; e++) {
                float val  = acc[i][j][e] + bv;
                size_t off = (size_t)(rowb + e) * N + col;
                if (BF16OUT) ((unsigned short*)outp)[off] = f2bf(val);
                else         ((float*)outp)[off] = val;
            }
        }
    }
}

// ---------- routed gather-attention: one wave32 per (b,h,s) ----------
__global__ __launch_bounds__(256) void routed_attention(
    const unsigned short* __restrict__ qkv,   // [B*S][3072] bf16
    const int* __restrict__ routes,           // [S][64]
    unsigned short* __restrict__ outa) {      // [B*S][1024] bf16
    __shared__ float qsh[8][64];
    __shared__ float ash[8][64];
    __shared__ int   rsh[8][64];

    const int tid  = threadIdx.x;
    const int lane = tid & 31;
    const int wv   = tid >> 5;
    const int w    = blockIdx.x * 8 + wv;
    const int b    = w >> 15;          // H*S = 32768
    const int rem  = w & 32767;
    const int h    = rem >> 11;        // S = 2048
    const int s    = rem & 2047;

    const size_t rowQ = (size_t)(b * Sv + s) * TDQ + h * HDv;
    qsh[wv][lane]      = bf2f(qkv[rowQ + lane]);
    qsh[wv][lane + 32] = bf2f(qkv[rowQ + lane + 32]);

    int r0 = routes[s * Kr + lane];
    int r1 = routes[s * Kr + lane + 32];
    r0 = r0 < 0 ? 0 : (r0 > Sv - 1 ? Sv - 1 : r0);
    r1 = r1 < 0 ? 0 : (r1 > Sv - 1 ? Sv - 1 : r1);
    rsh[wv][lane]      = r0;
    rsh[wv][lane + 32] = r1;
    __syncthreads();

    // scores: each lane owns route slots lane and lane+32
    float sc0 = 0.f, sc1 = 0.f;
    {
        const unsigned short* kp0 = qkv + (size_t)(b * Sv + r0) * TDQ + Dv + h * HDv;
        const unsigned short* kp1 = qkv + (size_t)(b * Sv + r1) * TDQ + Dv + h * HDv;
#pragma unroll
        for (int c = 0; c < 8; c++) {
            U4S u0, u1;
            u0.q = *(const uint4*)(kp0 + c * 8);
            u1.q = *(const uint4*)(kp1 + c * 8);
#pragma unroll
            for (int j = 0; j < 8; j++) {
                float qv = qsh[wv][c * 8 + j];
                sc0 += qv * bf2f(u0.s[j]);
                sc1 += qv * bf2f(u1.s[j]);
            }
        }
    }
    sc0 *= 0.125f;   // 1/sqrt(HD=64)
    sc1 *= 0.125f;

    // wave32 softmax over 64 slots
    float mx = fmaxf(sc0, sc1);
#pragma unroll
    for (int off = 16; off >= 1; off >>= 1)
        mx = fmaxf(mx, __shfl_xor(mx, off, 32));
    float e0 = __expf(sc0 - mx), e1 = __expf(sc1 - mx);
    float sm = e0 + e1;
#pragma unroll
    for (int off = 16; off >= 1; off >>= 1)
        sm += __shfl_xor(sm, off, 32);
    float inv = 1.0f / sm;
    ash[wv][lane]      = e0 * inv;
    ash[wv][lane + 32] = e1 * inv;
    __syncthreads();

    // out[d] = sum_k attn[k] * v[r_k][d]; each lane owns dims lane, lane+32
    float o0 = 0.f, o1 = 0.f;
    const int d0 = lane, d1 = lane + 32;
#pragma unroll 4
    for (int k = 0; k < 64; k++) {
        float a  = ash[wv][k];
        int   rv = rsh[wv][k];
        const unsigned short* vp =
            qkv + (size_t)(b * Sv + rv) * TDQ + 2 * Dv + h * HDv;
        o0 += a * bf2f(vp[d0]);
        o1 += a * bf2f(vp[d1]);
    }
    const size_t ob = (size_t)(b * Sv + s) * Dv + h * HDv;
    outa[ob + d0] = f2bf(o0);
    outa[ob + d1] = f2bf(o1);
}

extern "C" void kernel_launch(void* const* d_in, const int* in_sizes, int n_in,
                              void* d_out, int out_size, void* d_ws, size_t ws_size,
                              hipStream_t stream) {
    const float* x     = (const float*)d_in[0];
    const float* Wqkv  = (const float*)d_in[1];
    const float* bqkv  = (const float*)d_in[2];
    const float* Wproj = (const float*)d_in[3];
    const float* bproj = (const float*)d_in[4];
    const int*   routes = (const int*)d_in[5];
    float* out = (float*)d_out;

    char* ws = (char*)d_ws;
    size_t off = 0;
    unsigned short* xb     = (unsigned short*)(ws + off); off += (size_t)Bv * Sv * Dv * 2;
    unsigned short* wqkvb  = (unsigned short*)(ws + off); off += (size_t)Dv * TDQ * 2;
    unsigned short* wprojb = (unsigned short*)(ws + off); off += (size_t)Dv * Dv * 2;
    unsigned short* qkvb   = (unsigned short*)(ws + off); off += (size_t)Bv * Sv * TDQ * 2;
    unsigned short* attnb  = (unsigned short*)(ws + off);

    const int nx = Bv * Sv * Dv;       // 4,194,304
    const int nq = Dv * TDQ;           // 3,145,728
    const int np = Dv * Dv;            // 1,048,576
    cvt_f32_bf16<<<nx / 256, 256, 0, stream>>>(x,     xb,     nx);
    cvt_f32_bf16<<<nq / 256, 256, 0, stream>>>(Wqkv,  wqkvb,  nq);
    cvt_f32_bf16<<<np / 256, 256, 0, stream>>>(Wproj, wprojb, np);

    // QKV projection: [4096,1024] x [1024,3072] -> qkvb (bf16)
    dim3 g1(TDQ / 128, (Bv * Sv) / 128);
    gemm_bf16_wmma<true><<<g1, 256, 0, stream>>>(xb, wqkvb, bqkv, (void*)qkvb, TDQ, Dv);

    // routed attention: B*H*S waves, 8 waves per block
    routed_attention<<<(Bv * Hv * Sv) / 8, 256, 0, stream>>>(qkvb, routes, attnb);

    // output projection: [4096,1024] x [1024,1024] -> d_out (fp32)
    dim3 g2(Dv / 128, (Bv * Sv) / 128);
    gemm_bf16_wmma<false><<<g2, 256, 0, stream>>>(attnb, wprojb, bproj, (void*)out, Dv, Dv);
}
